// LSTMSequence_14328010899959
// MI455X (gfx1250) — matrix operand order repaired and checked by reading, not compiled, not run on hardware
//
#include <hip/hip_runtime.h>

typedef unsigned short u16;
typedef __attribute__((ext_vector_type(16))) __bf16 v16bf;
typedef __attribute__((ext_vector_type(8)))  float  v8f;

#define SEQ 4096
#define FEAT 2048
#define HID 2048
#define G4 8192   // 4*HID

// ---------- helpers ----------
__device__ inline u16 f32_to_bf16(float f) {
  unsigned u = __float_as_uint(f);
  u += 0x7fffu + ((u >> 16) & 1u);   // round-to-nearest-even
  return (u16)(u >> 16);
}
__device__ inline void unpack2(unsigned u, float& a, float& b) {
  a = __uint_as_float(u << 16);
  b = __uint_as_float(u & 0xffff0000u);
}
__device__ inline float sigmoidf_(float x) { return 1.0f / (1.0f + __expf(-x)); }

// 32-bit LDS byte offset of a __shared__ object (addrspacecast generic -> AS3)
__device__ __forceinline__ unsigned lds_addr_of(const void* p) {
  return (unsigned)(unsigned long long)(const __attribute__((address_space(3))) void*)p;
}

// CDNA5 async global->LDS copy, 16 bytes per lane, tracked on ASYNCcnt.
__device__ __forceinline__ void async_load_b128(unsigned lds_byte_off, const void* gptr) {
  asm volatile("global_load_async_to_lds_b128 %0, %1, off"
               :: "v"(lds_byte_off), "v"((unsigned long long)(uintptr_t)gptr)
               : "memory");
}
#define ASYNC_WAIT_8() asm volatile("s_wait_asynccnt 0x8" ::: "memory")
#define ASYNC_WAIT_0() asm volatile("s_wait_asynccnt 0x0" ::: "memory")

// ---------- 0) f32 -> bf16 convert ----------
__global__ __launch_bounds__(256)
void cvt_bf16_kernel(const float* __restrict__ in, u16* __restrict__ out, int n) {
  int i = blockIdx.x * 256 + threadIdx.x;
  if (i < n) out[i] = f32_to_bf16(in[i]);
}

// ---------- 1) xg = x @ W_ih^T + (b_ih + b_hh) ----------
// Block tile 128(M) x 128(N), K-panels of 64, double-buffered in LDS via
// GLOBAL_LOAD_ASYNC_TO_LDS_B128; 8 waves, each wave owns a 32x64 tile
// (2x4 WMMA 16x16 accumulators), v_wmma_f32_16x16x32_bf16 with f32 accum.
#define BM 128
#define BN 128
#define BK 64

union AFrag { uint4 u[2]; v16bf v; };

__global__ __launch_bounds__(256)
void gemm_xg_kernel(const u16* __restrict__ Abf,   // [SEQ, FEAT] bf16, k-major
                    const u16* __restrict__ Bbf,   // [G4, FEAT]  bf16, k-major (W_ih)
                    const float* __restrict__ b_ih,
                    const float* __restrict__ b_hh,
                    float* __restrict__ xg)        // [SEQ, G4] f32
{
  __shared__ __align__(16) u16 ldsA[2][BM * BK];   // 2 x 16 KB
  __shared__ __align__(16) u16 ldsB[2][BN * BK];   // 2 x 16 KB

  const int tid  = threadIdx.x;
  const int lane = tid & 31;
  const int wave = tid >> 5;                 // 8 waves
  const int waveM = wave >> 1;               // 0..3 -> 32-row strip
  const int waveN = wave & 1;                // 0..1 -> 64-col strip
  const int m0 = blockIdx.x * BM;            // gridDim.x = SEQ/BM = 32
  const int n0 = blockIdx.y * BN;            // gridDim.y = G4/BN  = 64
  const bool hi = lane >= 16;
  const int l15 = lane & 15;

  v8f acc[2][4] = {};

  // Issue one K-panel's async loads: 256 threads x (4 A + 4 B) 16B chunks.
  auto loadPanel = [&](int p, int buf) {
    const int kp = p * BK;
#pragma unroll
    for (int i = 0; i < 4; ++i) {
      const int c = tid + i * 256;           // chunk id 0..1023
      const int m = c >> 3;                  // row 0..127
      const int kc = (c & 7) * 8;            // 8 chunks of 8 bf16 per 64-elem row
      async_load_b128(lds_addr_of(&ldsA[buf][m * BK + kc]),
                      Abf + (size_t)(m0 + m) * FEAT + kp + kc);
      async_load_b128(lds_addr_of(&ldsB[buf][m * BK + kc]),
                      Bbf + (size_t)(n0 + m) * FEAT + kp + kc);
    }
  };

  const int numPanels = FEAT / BK;           // 32
  loadPanel(0, 0);

  for (int p = 0; p < numPanels; ++p) {
    const int buf = p & 1;
    if (p + 1 < numPanels) {
      loadPanel(p + 1, buf ^ 1);             // prefetch next panel
      ASYNC_WAIT_8();                        // panel p's 8 ops retired (in-order)
    } else {
      ASYNC_WAIT_0();
    }
    __syncthreads();                         // all waves' panel-p data in LDS

#pragma unroll
    for (int kk = 0; kk < BK; kk += 32) {
      // A fragments (16x32): lanes 0-15 hold K {0..7,16..23}, lanes 16-31 {8..15,24..31}
      AFrag a[2];
#pragma unroll
      for (int ms = 0; ms < 2; ++ms) {
        const u16* ab = &ldsA[buf][(waveM * 32 + ms * 16 + l15) * BK + kk + (hi ? 8 : 0)];
        a[ms].u[0] = *(const uint4*)ab;
        a[ms].u[1] = *(const uint4*)(ab + 16);
      }
      // B fragments (32x16): lane = col, lanes 0-15 hold K {0..7,16..23}, lanes 16-31 {8..15,24..31}
      v16bf b[4];
#pragma unroll
      for (int ns = 0; ns < 4; ++ns) {
        const u16* bb = &ldsB[buf][(waveN * 64 + ns * 16 + l15) * BK + kk + (hi ? 8 : 0)];
        AFrag bf;
        bf.u[0] = *(const uint4*)bb;
        bf.u[1] = *(const uint4*)(bb + 16);
        b[ns] = bf.v;
      }
#pragma unroll
      for (int ms = 0; ms < 2; ++ms)
#pragma unroll
        for (int ns = 0; ns < 4; ++ns)
          acc[ms][ns] = __builtin_amdgcn_wmma_f32_16x16x32_bf16(
              false, a[ms].v, false, b[ns], (short)0, acc[ms][ns], false, false);
    }
    __syncthreads();                         // done reading buf before it is refilled
  }

#pragma unroll
  for (int ms = 0; ms < 2; ++ms) {
#pragma unroll
    for (int ns = 0; ns < 4; ++ns) {
      const int n = n0 + waveN * 64 + ns * 16 + l15;
      const float bias = b_ih[n] + b_hh[n];
#pragma unroll
      for (int v = 0; v < 8; ++v) {
        const int m = m0 + waveM * 32 + ms * 16 + v + (hi ? 8 : 0);
        xg[(size_t)m * G4 + n] = acc[ms][ns][v] + bias;
      }
    }
  }
}

// ---------- 2) persistent recurrent scan: 16 blocks x 256 threads ----------
__global__ __launch_bounds__(256)
void lstm_scan_kernel(const float* __restrict__ xg,   // [SEQ, G4]
                      const u16* __restrict__ Whh,    // [G4, HID] bf16 (L2-resident: 33.5 MB)
                      u16* __restrict__ hbf,          // [HID] bf16 (zero-init by memset)
                      float* __restrict__ hf32,       // [HID] f32
                      int* __restrict__ bar)          // zero-init by memset
{
  __shared__ __align__(16) unsigned h_lds[HID / 2];   // 2048 bf16 packed as 1024 u32
  __shared__ float gates[512];
  __shared__ float c_lds[128];

  const int tid = threadIdx.x;
  const int wg = blockIdx.x;                 // 0..15, owns hidden units [wg*128, wg*128+128)
  if (tid < 128) c_lds[tid] = 0.0f;

  const int r0 = tid, r1 = tid + 256;        // local gate rows 0..511
  const int g0 = r0 >> 7, j0 = r0 & 127;
  const int g1 = r1 >> 7, j1 = r1 & 127;
  const size_t row0 = (size_t)g0 * HID + wg * 128 + j0;
  const size_t row1 = (size_t)g1 * HID + wg * 128 + j1;
  const u16* w0p = Whh + row0 * HID;
  const u16* w1p = Whh + row1 * HID;

  for (int t = 0; t < SEQ; ++t) {
    // broadcast h into LDS via async DMA: 256 lanes x 16 B = 4 KB
    async_load_b128(lds_addr_of(&h_lds[tid * 4]), hbf + tid * 8);
    ASYNC_WAIT_0();
    __syncthreads();

    const float* xgt = xg + (size_t)t * G4;
    if (t + 1 < SEQ) __builtin_prefetch(xg + (size_t)(t + 1) * G4 + row0, 0, 0);

    float acc0 = xgt[row0];
    float acc1 = xgt[row1];
    for (int k = 0; k < HID; k += 8) {
      uint4 hv = *((const uint4*)h_lds + (k >> 3));
      uint4 wa = *(const uint4*)(w0p + k);
      uint4 wb = *(const uint4*)(w1p + k);
      float h0,h1,h2,h3,h4,h5,h6,h7;
      unpack2(hv.x, h0, h1); unpack2(hv.y, h2, h3);
      unpack2(hv.z, h4, h5); unpack2(hv.w, h6, h7);
      float a0,a1,a2,a3,a4,a5,a6,a7;
      unpack2(wa.x, a0, a1); unpack2(wa.y, a2, a3);
      unpack2(wa.z, a4, a5); unpack2(wa.w, a6, a7);
      float b0,b1,b2,b3,b4,b5,b6,b7;
      unpack2(wb.x, b0, b1); unpack2(wb.y, b2, b3);
      unpack2(wb.z, b4, b5); unpack2(wb.w, b6, b7);
      acc0 = fmaf(a0,h0, fmaf(a1,h1, fmaf(a2,h2, fmaf(a3,h3,
             fmaf(a4,h4, fmaf(a5,h5, fmaf(a6,h6, fmaf(a7,h7, acc0))))))));
      acc1 = fmaf(b0,h0, fmaf(b1,h1, fmaf(b2,h2, fmaf(b3,h3,
             fmaf(b4,h4, fmaf(b5,h5, fmaf(b6,h6, fmaf(b7,h7, acc1))))))));
    }
    gates[r0] = acc0;
    gates[r1] = acc1;
    __syncthreads();

    if (tid < 128) {                         // update 128 hidden units owned by this WG
      float ig = sigmoidf_(gates[tid]);
      float fg = sigmoidf_(gates[128 + tid]);
      float gg = tanhf(gates[256 + tid]);
      float og = sigmoidf_(gates[384 + tid]);
      float c = fg * c_lds[tid] + ig * gg;
      c_lds[tid] = c;
      float h = og * tanhf(c);
      int j = wg * 128 + tid;
      hbf[j] = f32_to_bf16(h);
      hf32[j] = h;
    }
    __syncthreads();

    // grid barrier across the 16 persistent blocks
    if (tid == 0) {
      __threadfence();
      atomicAdd(bar, 1);
      const int target = 16 * (t + 1);
      while (__hip_atomic_load(bar, __ATOMIC_ACQUIRE, __HIP_MEMORY_SCOPE_AGENT) < target) {
        __builtin_amdgcn_s_sleep(1);
      }
    }
    __syncthreads();
  }
}

// ---------- 3) out = sigmoid(W_lin @ h + b_lin) ----------
__global__ __launch_bounds__(256)
void final_linear_kernel(const float* __restrict__ hf32,
                         const float* __restrict__ W_lin,
                         const float* __restrict__ b_lin,
                         float* __restrict__ out)
{
  __shared__ float red[256];
  float s = 0.0f;
  for (int j = threadIdx.x; j < HID; j += 256) s += W_lin[j] * hf32[j];
  red[threadIdx.x] = s;
  __syncthreads();
  for (int k = 128; k > 0; k >>= 1) {
    if (threadIdx.x < k) red[threadIdx.x] += red[threadIdx.x + k];
    __syncthreads();
  }
  if (threadIdx.x == 0) out[0] = sigmoidf_(red[0] + b_lin[0]);
}

// ---------- launcher ----------
extern "C" void kernel_launch(void* const* d_in, const int* in_sizes, int n_in,
                              void* d_out, int out_size, void* d_ws, size_t ws_size,
                              hipStream_t stream) {
  const float* x     = (const float*)d_in[0];   // [1, SEQ, FEAT]
  const float* W_ih  = (const float*)d_in[1];   // [G4, FEAT]
  const float* W_hh  = (const float*)d_in[2];   // [G4, HID]
  const float* b_ih  = (const float*)d_in[3];   // [G4]
  const float* b_hh  = (const float*)d_in[4];   // [G4]
  const float* W_lin = (const float*)d_in[5];   // [1, HID]
  const float* b_lin = (const float*)d_in[6];   // [1]
  float* out = (float*)d_out;

  // workspace layout
  char* base = (char*)d_ws;
  size_t off = 0;
  u16* xbf   = (u16*)(base + off); off += (size_t)SEQ * FEAT * sizeof(u16);   // 16.8 MB
  u16* wihbf = (u16*)(base + off); off += (size_t)G4  * FEAT * sizeof(u16);   // 33.5 MB
  u16* whhbf = (u16*)(base + off); off += (size_t)G4  * HID  * sizeof(u16);   // 33.5 MB
  float* xg  = (float*)(base + off); off += (size_t)SEQ * G4 * sizeof(float); // 134.2 MB
  u16* hbf   = (u16*)(base + off); off += 4096;                                // h bf16 (padded)
  float* hf32= (float*)(base + off); off += HID * sizeof(float);
  int* bar   = (int*)(base + off); off += 64;

  // 0) convert to bf16
  {
    int n0 = SEQ * FEAT;
    cvt_bf16_kernel<<<(n0 + 255) / 256, 256, 0, stream>>>(x, xbf, n0);
    int n1 = G4 * FEAT;
    cvt_bf16_kernel<<<(n1 + 255) / 256, 256, 0, stream>>>(W_ih, wihbf, n1);
    int n2 = G4 * HID;
    cvt_bf16_kernel<<<(n2 + 255) / 256, 256, 0, stream>>>(W_hh, whhbf, n2);
  }

  // 1) big input-side GEMM via WMMA + async-LDS double buffering
  gemm_xg_kernel<<<dim3(SEQ / BM, G4 / BN), 256, 0, stream>>>(xbf, wihbf, b_ih, b_hh, xg);

  // 2) zero-init h and barrier counter (graph-capture safe), then persistent scan
  hipMemsetAsync(hbf, 0, HID * sizeof(u16), stream);
  hipMemsetAsync(bar, 0, 64, stream);
  lstm_scan_kernel<<<16, 256, 0, stream>>>(xg, whhbf, hbf, hf32, bar);

  // 3) final linear + sigmoid
  final_linear_kernel<<<1, 256, 0, stream>>>(hf32, W_lin, b_lin, out);
}
